// Model_78245714198585
// MI455X (gfx1250) — matrix operand order repaired and checked by reading, not compiled, not run on hardware
//
#include <hip/hip_runtime.h>
#include <hip/hip_bf16.h>

// ---------------------------------------------------------------------------
// CDNA5 (gfx1250, wave32) implementation of the temporal-attention model.
// All dense contractions run through v_wmma_f32_16x16x32_bf16, fed by
// contiguous ds_load_b128 operand reads (k-major LDS staging).
// ---------------------------------------------------------------------------

typedef __bf16 bf16;
typedef bf16  v16bf __attribute__((ext_vector_type(16)));
typedef bf16  v8bf  __attribute__((ext_vector_type(8)));
typedef float v8f   __attribute__((ext_vector_type(8)));
typedef float v4f   __attribute__((ext_vector_type(4)));

#define T_LEN 8192
#define BATCH 8
#define NFEAT 256
#define MROWS (BATCH * T_LEN)   // 65536 token rows

#define WMMA_BF16(a, b, c) \
  __builtin_amdgcn_wmma_f32_16x16x32_bf16(false, (a), false, (b), (short)0, (c), false, false)

__device__ __forceinline__ v16bf cat8(v8bf lo, v8bf hi) {
  return __builtin_shufflevector(lo, hi, 0, 1, 2, 3, 4, 5, 6, 7,
                                          8, 9, 10, 11, 12, 13, 14, 15);
}

// Per CDNA5 ISA 7.12.2 (16-bit operands, wave32):
//  A 16x32: lane<16 -> M=lane, K chunks {0..7,16..23}; lane>=16 -> M=lane-16, {8..15,24..31}
//  B 32x16: lane<16 -> N=lane, K chunks {0..7,16..23}; lane>=16 -> N=lane-16, {8..15,24..31}
// With k-contiguous LDS rows these are 2x ds_load_b128 per operand.
__device__ __forceinline__ v16bf loadA(const bf16* rowp, int lane) {
  int kbA = (lane & 16) ? 8 : 0;
  v8bf lo = *(const v8bf*)(rowp + kbA);
  v8bf hi = *(const v8bf*)(rowp + kbA + 16);
  return cat8(lo, hi);
}
__device__ __forceinline__ v16bf loadB(const bf16* rowp, int lane) {
  int kbB = (lane & 16) ? 8 : 0;
  v8bf lo = *(const v8bf*)(rowp + kbB);
  v8bf hi = *(const v8bf*)(rowp + kbB + 16);
  return cat8(lo, hi);
}

__device__ __forceinline__ v8bf pack8(v4f f0, v4f f1) {
  v8bf p;
#pragma unroll
  for (int u = 0; u < 4; u++) { p[u] = (bf16)f0[u]; p[u + 4] = (bf16)f1[u]; }
  return p;
}

__device__ __forceinline__ float act_apply(float v, int act) {
  if (act == 1) return fmaxf(v, 0.f);                                    // relu
  if (act == 2) return 1.f / (1.f + expf(-v));                           // sigmoid
  if (act == 3) return 0.5f * v * (1.f + erff(v * 0.7071067811865476f)); // exact gelu
  return v;
}

// ---------------------------------------------------------------------------
// Generic token-major WMMA GEMM:  C[row][n] = epi( sum_k A'[row][k]*Bw[n][k] + bias[n] )
// Bw is bf16 n-major [n][K].
// MODE 0 (PLAIN): kc == K (mult of 32), rows always valid, no guards.
// MODE 1 (CONV3): kc = 256, K = 768; logical k = j*256 + c, row shifted by
//                 (j-1)*dil within its batch (zero pad outside).
// MODE 2 (K19):   K = kc = 19, single zero-padded k-block.
// bStride != 0 selects a per-batch B matrix.  Epilogue: act, optional mask
// (pre/post residual add), optional residual add.
// Block: 256 thr = 8 waves; C macro-tile 128x64; each wave -> 2x2 16x16 tiles.
// ---------------------------------------------------------------------------
template <int MODE>
__global__ __launch_bounds__(256)
void k_gemm(const float* __restrict__ A, const bf16* __restrict__ Bw,
            const float* __restrict__ bias, const float* __restrict__ addp,
            const float* __restrict__ maskp, float* __restrict__ C,
            int N, int K, int dil, int act, int maskMode, long long bStride) {
  __shared__ __align__(16) bf16 AsT[128][40];  // [m][k], 80B row stride
  __shared__ __align__(16) bf16 BsT[64][40];   // [n][k]
  const int tid      = threadIdx.x;
  const int rowBase  = blockIdx.x * 128;
  const int colBase  = blockIdx.y * 64;
  const int bidx     = rowBase / T_LEN;
  const int trowBase = rowBase - bidx * T_LEN;
  const bf16* Bp = Bw + (long long)bidx * bStride;
  const int wave = tid >> 5, lane = tid & 31;
  const int wr = wave >> 1, wc = wave & 1;     // 4x2 wave grid of 32x32 blocks
  const int lmA = tid >> 1, lkA = (tid & 1) * 16;   // A loader: 16 k per thread
  const int lmB = tid >> 2, lkB = (tid & 3) * 8;    // B loader: 8 k per thread
  v8f acc[2][2] = {{{}, {}}, {{}, {}}};

  for (int kb = 0; kb < K; kb += 32) {
    if (MODE == 0) {            // ---- A tile, no guards ----
      const float* src = A + (long long)(rowBase + lmA) * K + kb + lkA;
      *(v8bf*)&AsT[lmA][lkA]     = pack8(*(const v4f*)src, *(const v4f*)(src + 4));
      *(v8bf*)&AsT[lmA][lkA + 8] = pack8(*(const v4f*)(src + 8), *(const v4f*)(src + 12));
      if (kb + 32 < K) __builtin_prefetch(src + 32, 0, 0);
    } else if (MODE == 1) {     // ---- A tile, conv3 row shift ----
      int j = kb >> 8;
      int cb = kb & 255;
      int t = trowBase + lmA + (j - 1) * dil;
      v8bf p0 = {}, p1 = {};
      if (t >= 0 && t < T_LEN) {
        const float* src = A + ((long long)(bidx * T_LEN + t)) * 256 + cb + lkA;
        p0 = pack8(*(const v4f*)src, *(const v4f*)(src + 4));
        p1 = pack8(*(const v4f*)(src + 8), *(const v4f*)(src + 12));
      }
      *(v8bf*)&AsT[lmA][lkA]     = p0;
      *(v8bf*)&AsT[lmA][lkA + 8] = p1;
    } else {                    // ---- A tile, K = 19 zero padded ----
      const float* src = A + (long long)(rowBase + lmA) * 19;
      v8bf p0 = {}, p1 = {};
#pragma unroll
      for (int u = 0; u < 8; u++) {
        int c0 = lkA + u, c1 = lkA + 8 + u;
        if (c0 < 19) p0[u] = (bf16)src[c0];
        if (c1 < 19) p1[u] = (bf16)src[c1];
      }
      *(v8bf*)&AsT[lmA][lkA]     = p0;
      *(v8bf*)&AsT[lmA][lkA + 8] = p1;
    }
    {   // ---- B tile: n-major global, straight copy ----
      int gn = colBase + lmB;
      v8bf pk = {};
      const bf16* srcB = Bp + (long long)gn * K + kb + lkB;
      if (MODE != 2) {
        if (gn < N) pk = *(const v8bf*)srcB;
      } else {
#pragma unroll
        for (int u = 0; u < 8; u++) {
          int gk = kb + lkB + u;
          if (gn < N && gk < 19) pk[u] = srcB[u];
        }
      }
      *(v8bf*)&BsT[lmB][lkB] = pk;
    }
    __syncthreads();
    const int la = lane & 15;
    v16bf a0 = loadA(&AsT[wr * 32 + la][0], lane);
    v16bf a1 = loadA(&AsT[wr * 32 + 16 + la][0], lane);
    v16bf b0 = loadB(&BsT[wc * 32 + la][0], lane);
    v16bf b1 = loadB(&BsT[wc * 32 + 16 + la][0], lane);
    acc[0][0] = WMMA_BF16(a0, b0, acc[0][0]);
    acc[0][1] = WMMA_BF16(a0, b1, acc[0][1]);
    acc[1][0] = WMMA_BF16(a1, b0, acc[1][0]);
    acc[1][1] = WMMA_BF16(a1, b1, acc[1][1]);
    __syncthreads();
  }

  const int rhalf = (lane & 16) ? 8 : 0;
#pragma unroll
  for (int ti = 0; ti < 2; ti++) {
#pragma unroll
    for (int tj = 0; tj < 2; tj++) {
      int gc = colBase + wc * 32 + tj * 16 + (lane & 15);
      if (gc >= N) continue;
      float bval = bias ? bias[gc] : 0.f;
#pragma unroll
      for (int r = 0; r < 8; r++) {
        long long gr = rowBase + wr * 32 + ti * 16 + rhalf + r;
        float v = acc[ti][tj][r] + bval;
        v = act_apply(v, act);
        float mk = maskp ? maskp[gr] : 1.f;
        if (maskMode == 2) v *= mk;                 // mask before residual add
        if (addp) v += addp[gr * (long long)N + gc];
        if (maskMode == 1) v *= mk;                 // mask after residual add
        C[gr * (long long)N + gc] = v;
      }
    }
  }
}

// ---- kv[b][h][d][e] = sum_l K[l,d] * V[l,e]  (K-split deterministic) -------
__global__ __launch_bounds__(256)
void k_kv_part(const float* __restrict__ KS, const float* __restrict__ VS,
               float* __restrict__ KVP) {
  __shared__ __align__(16) bf16 KsT[64][40];  // [d][token]
  __shared__ __align__(16) bf16 VsT[64][40];  // [e][token]
  const int ks = blockIdx.x;            // 0..7 K-split
  const int bh = blockIdx.y;            // 0..31
  const int b = bh >> 2, h = bh & 3;
  const int tid = threadIdx.x;
  const int wave = tid >> 5, lane = tid & 31;
  const int tm = wave >> 1, tn0 = (wave & 1) * 2;
  const int lkk = tid & 31;             // token within slab
  const int ld0 = (tid >> 5) * 8;       // 8 consecutive channels
  v8f acc0 = {}; v8f acc1 = {};
  for (int it = 0; it < 32; it++) {
    int l0 = ks * 1024 + it * 32;
    long long base = ((long long)(b * T_LEN + l0 + lkk)) * NFEAT + h * 64 + ld0;
    v4f k0 = *(const v4f*)(KS + base);
    v4f k1 = *(const v4f*)(KS + base + 4);
    v4f q0 = *(const v4f*)(VS + base);
    v4f q1 = *(const v4f*)(VS + base + 4);
#pragma unroll
    for (int u = 0; u < 4; u++) {
      KsT[ld0 + u][lkk]     = (bf16)k0[u];
      KsT[ld0 + 4 + u][lkk] = (bf16)k1[u];
      VsT[ld0 + u][lkk]     = (bf16)q0[u];
      VsT[ld0 + 4 + u][lkk] = (bf16)q1[u];
    }
    __syncthreads();
    int mrow = tm * 16 + (lane & 15);
    int n0   = tn0 * 16 + (lane & 15);
    v16bf av  = loadA(&KsT[mrow][0], lane);
    v16bf bv0 = loadB(&VsT[n0][0], lane);
    v16bf bv1 = loadB(&VsT[n0 + 16][0], lane);
    acc0 = WMMA_BF16(av, bv0, acc0);
    acc1 = WMMA_BF16(av, bv1, acc1);
    __syncthreads();
  }
  int rsub = tm * 16 + ((lane & 16) ? 8 : 0);
  long long base = ((long long)(bh * 8 + ks)) * 4096;
  for (int tt = 0; tt < 2; tt++) {
    int gc = (tn0 + tt) * 16 + (lane & 15);
    v8f acc = tt ? acc1 : acc0;
#pragma unroll
    for (int r = 0; r < 8; r++)
      KVP[base + (rsub + r) * 64 + gc] = acc[r];
  }
}

// reduce K-split partials; emit kv pre-transposed in bf16: KVT[bh][e*64+d]
__global__ void k_kv_reduce(const float* __restrict__ KVP, bf16* __restrict__ KVT) {
  int idx = blockIdx.x * 256 + threadIdx.x;      // 32*4096 outputs
  int bh = idx >> 12;
  int e = (idx >> 6) & 63, d = idx & 63;
  float s = 0.f;
  for (int ks = 0; ks < 8; ks++)
    s += KVP[((long long)(bh * 8 + ks)) * 4096 + d * 64 + e];
  KVT[idx] = (bf16)s;
}

// ---- ATT[row][h*64+e] = z[row,h] * sum_d Q[row,d] * kv[d,e] ----------------
__global__ __launch_bounds__(256)
void k_qkvz(const float* __restrict__ QS, const bf16* __restrict__ KVT,
            const float* __restrict__ Z, float* __restrict__ ATT) {
  __shared__ __align__(16) bf16 Qs[64][72];   // [token][d], 144B row stride
  __shared__ __align__(16) bf16 Kv[64][72];   // [e][d]
  const int tid = threadIdx.x;
  const int rowBase = blockIdx.x * 64;
  const int h = blockIdx.y;
  const int b = rowBase / T_LEN;
  {
    int m = tid >> 2, d0 = (tid & 3) * 16;
    const float* pq = QS + ((long long)(rowBase + m)) * NFEAT + h * 64 + d0;
    *(v8bf*)&Qs[m][d0]     = pack8(*(const v4f*)pq, *(const v4f*)(pq + 4));
    *(v8bf*)&Qs[m][d0 + 8] = pack8(*(const v4f*)(pq + 8), *(const v4f*)(pq + 12));
    const bf16* pk = KVT + ((long long)(b * 4 + h)) * 4096 + m * 64 + d0;
    *(v8bf*)&Kv[m][d0]     = *(const v8bf*)pk;
    *(v8bf*)&Kv[m][d0 + 8] = *(const v8bf*)(pk + 8);
  }
  __syncthreads();
  const int wave = tid >> 5, lane = tid & 31;
  const int tm = wave >> 1, tn0 = (wave & 1) * 2;
  v8f acc0 = {}; v8f acc1 = {};
#pragma unroll
  for (int kb = 0; kb < 64; kb += 32) {
    int mrow = tm * 16 + (lane & 15);
    int n0   = tn0 * 16 + (lane & 15);
    v16bf av  = loadA(&Qs[mrow][kb], lane);
    v16bf bv0 = loadB(&Kv[n0][kb], lane);
    v16bf bv1 = loadB(&Kv[n0 + 16][kb], lane);
    acc0 = WMMA_BF16(av, bv0, acc0);
    acc1 = WMMA_BF16(av, bv1, acc1);
  }
  int rsub = tm * 16 + ((lane & 16) ? 8 : 0);
  for (int tt = 0; tt < 2; tt++) {
    int gc = h * 64 + (tn0 + tt) * 16 + (lane & 15);
    v8f acc = tt ? acc1 : acc0;
#pragma unroll
    for (int r = 0; r < 8; r++) {
      long long gr = rowBase + rsub + r;
      ATT[gr * NFEAT + gc] = acc[r] * Z[gr * 4 + h];
    }
  }
}

// ---- SFI correlation partials: CP[b,ks][c][d] = sum_l FS[l,c]*FT[l,d] ------
__global__ __launch_bounds__(256)
void k_corr_part(const float* __restrict__ FS, const float* __restrict__ FT,
                 float* __restrict__ CP) {
  __shared__ __align__(16) bf16 AsT[64][40];  // [c][token]
  __shared__ __align__(16) bf16 BsT[64][40];  // [d][token]
  const int tileId = blockIdx.x;          // 0..15 (4x4 tiles of 64x64)
  const int ks = blockIdx.y;              // 0..7
  const int b  = blockIdx.z;              // 0..7
  const int cBase = (tileId >> 2) * 64, dBase = (tileId & 3) * 64;
  const int tid = threadIdx.x;
  const int wave = tid >> 5, lane = tid & 31;
  const int tm = wave >> 1, tn0 = (wave & 1) * 2;
  const int lkk = tid & 31;
  const int lc0 = (tid >> 5) * 8;
  v8f acc0 = {}; v8f acc1 = {};
  for (int it = 0; it < 32; it++) {
    int l0 = ks * 1024 + it * 32;
    long long row = ((long long)(b * T_LEN + l0 + lkk)) * NFEAT;
    v4f a0 = *(const v4f*)(FS + row + cBase + lc0);
    v4f a1 = *(const v4f*)(FS + row + cBase + lc0 + 4);
    v4f b0 = *(const v4f*)(FT + row + dBase + lc0);
    v4f b1 = *(const v4f*)(FT + row + dBase + lc0 + 4);
#pragma unroll
    for (int u = 0; u < 4; u++) {
      AsT[lc0 + u][lkk]     = (bf16)a0[u];
      AsT[lc0 + 4 + u][lkk] = (bf16)a1[u];
      BsT[lc0 + u][lkk]     = (bf16)b0[u];
      BsT[lc0 + 4 + u][lkk] = (bf16)b1[u];
    }
    __syncthreads();
    int mrow = tm * 16 + (lane & 15);
    int n0   = tn0 * 16 + (lane & 15);
    v16bf av  = loadA(&AsT[mrow][0], lane);
    v16bf bv0 = loadB(&BsT[n0][0], lane);
    v16bf bv1 = loadB(&BsT[n0 + 16][0], lane);
    acc0 = WMMA_BF16(av, bv0, acc0);
    acc1 = WMMA_BF16(av, bv1, acc1);
    __syncthreads();
  }
  int rsub = tm * 16 + ((lane & 16) ? 8 : 0);
  long long base = ((long long)(b * 8 + ks)) << 16;
  for (int tt = 0; tt < 2; tt++) {
    int gc = dBase + (tn0 + tt) * 16 + (lane & 15);
    v8f acc = tt ? acc1 : acc0;
#pragma unroll
    for (int r = 0; r < 8; r++)
      CP[base + (long long)(cBase + rsub + r) * 256 + gc] = acc[r];
  }
}

__global__ void k_corr_reduce(const float* __restrict__ CP, float* __restrict__ Msm) {
  int idx = blockIdx.x * 256 + threadIdx.x;      // 8*65536
  int b = idx >> 16, cd = idx & 65535;
  float s = 0.f;
  for (int ks = 0; ks < 8; ks++) s += CP[(((long long)(b * 8 + ks)) << 16) + cd];
  Msm[idx] = s * (1.f / (float)T_LEN);
}

__global__ void k_softmax(float* __restrict__ Msm) {
  int b = blockIdx.x >> 8, c = blockIdx.x & 255;
  float* rowp = Msm + (((long long)b) << 16) + (c << 8);
  int d = threadIdx.x;
  __shared__ float red[256];
  float v = rowp[d];
  red[d] = v; __syncthreads();
  for (int s = 128; s > 0; s >>= 1) { if (d < s) red[d] = fmaxf(red[d], red[d + s]); __syncthreads(); }
  float mx = red[0]; __syncthreads();
  float e = expf(v - mx);
  red[d] = e; __syncthreads();
  for (int s = 128; s > 0; s >>= 1) { if (d < s) red[d] += red[d + s]; __syncthreads(); }
  rowp[d] = e / red[0];
}

// m is consumed n-major by the cross GEMM (B[n=c][k=d] = m[c][d]): pure convert
__global__ void k_mt(const float* __restrict__ Msm, bf16* __restrict__ MT) {
  int idx = blockIdx.x * 256 + threadIdx.x;      // 8*65536
  MT[idx] = (bf16)Msm[idx];
}

// ---- conv_in: 64 -> 6 channels, split into 5 SFI feature maps + ft0 --------
__global__ void k_conv_in(const float* __restrict__ x, const float* __restrict__ w,
                          const float* __restrict__ bia, float* __restrict__ fsin,
                          float* __restrict__ ft0v) {
  long long idx = (long long)blockIdx.x * 256 + threadIdx.x;
  const long long tot = (long long)MROWS * 19;
  if (idx >= tot) return;
  int v = (int)(idx % 19);
  long long bt = idx / 19;
  int t = (int)(bt % T_LEN);
  int b = (int)(bt / T_LEN);
  float acc[6];
#pragma unroll
  for (int o = 0; o < 6; o++) acc[o] = bia[o];
  for (int c = 0; c < 64; c++) {
    float xv = x[(((long long)(b * 64 + c)) * T_LEN + t) * 19 + v];
#pragma unroll
    for (int o = 0; o < 6; o++) acc[o] += w[o * 64 + c] * xv;
  }
  long long base = bt * 19 + v;
  const long long SZ = (long long)MROWS * 19;
#pragma unroll
  for (int s = 0; s < 5; s++) fsin[s * SZ + base] = acc[s];
  ft0v[base] = acc[5];
}

// ---- deterministic two-stage per-(b,c) token reductions --------------------
__global__ void k_stats1(const float* __restrict__ X, float* __restrict__ PS,
                         float* __restrict__ PS2) {
  int b = blockIdx.x >> 5, ch = blockIdx.x & 31;
  int c = threadIdx.x;
  float s = 0.f, s2 = 0.f;
  for (int it = 0; it < 256; it++) {
    float v = X[((long long)(b * T_LEN + ch * 256 + it)) * NFEAT + c];
    s += v; s2 += v * v;
  }
  PS[blockIdx.x * 256 + c] = s;
  PS2[blockIdx.x * 256 + c] = s2;
}

__global__ void k_stats2(const float* __restrict__ PS, const float* __restrict__ PS2,
                         float* __restrict__ MU, float* __restrict__ RS,
                         float* __restrict__ KSUM, int mode) {
  int b = blockIdx.x, c = threadIdx.x;
  float s = 0.f, s2 = 0.f;
  for (int ch = 0; ch < 32; ch++) {
    s += PS[(b * 32 + ch) * 256 + c];
    s2 += PS2[(b * 32 + ch) * 256 + c];
  }
  if (mode == 1) {
    float mu = s * (1.f / (float)T_LEN);
    float var = s2 * (1.f / (float)T_LEN) - mu * mu;
    MU[b * 256 + c] = mu;
    RS[b * 256 + c] = rsqrtf(var + 1e-5f);
  } else {
    KSUM[b * 256 + c] = s;
  }
}

__global__ void k_norm(const float* __restrict__ X, const float* __restrict__ MU,
                       const float* __restrict__ RS, float* __restrict__ Y) {
  long long idx = (long long)blockIdx.x * 256 + threadIdx.x;
  int c = (int)(idx & 255);
  int row = (int)(idx >> 8);
  int b = row >> 13;
  Y[idx] = (X[idx] - MU[b * 256 + c]) * RS[b * 256 + c];
}

__global__ void k_z(const float* __restrict__ QS, const float* __restrict__ KSUM,
                    float* __restrict__ Z) {
  int idx = blockIdx.x * 256 + threadIdx.x;      // MROWS*4
  int h = idx & 3;
  long long row = idx >> 2;
  int b = (int)(row >> 13);
  float s = 0.f;
  for (int d = 0; d < 64; d++)
    s += QS[row * NFEAT + h * 64 + d] * (KSUM[b * 256 + h * 64 + d] + 1e-6f);
  Z[idx] = 1.f / s;
}

// ---- weight conversion: src[o][i][j] -> bf16 dst[o][j*I + i] (n-major) -----
__global__ void k_prep(const float* __restrict__ src, bf16* __restrict__ dst,
                       int O, int I, int J) {
  int idx = blockIdx.x * 256 + threadIdx.x;
  if (idx >= O * I * J) return;
  int o = idx / (I * J);
  int rem = idx - o * (I * J);
  int i = rem / J;
  int j = rem - i * J;
  dst[(long long)o * (I * J) + j * I + i] = (bf16)src[idx];
}

// ---- token-major (M x 48) -> (B,48,T) with mask ----------------------------
__global__ void k_out(const float* __restrict__ CT, const float* __restrict__ mask,
                      float* __restrict__ out) {
  long long idx = (long long)blockIdx.x * 256 + threadIdx.x;  // MROWS*48
  long long row = idx / 48;
  int n = (int)(idx - row * 48);
  int b = (int)(row >> 13);
  int t = (int)(row & (T_LEN - 1));
  out[(((long long)b * 48 + n) << 13) + t] = CT[idx] * mask[row];
}

// ---------------------------------------------------------------------------
extern "C" void kernel_launch(void* const* d_in, const int* in_sizes, int n_in,
                              void* d_out, int out_size, void* d_ws, size_t ws_size,
                              hipStream_t stream) {
  (void)in_sizes; (void)n_in; (void)out_size; (void)ws_size;
  const float* x     = (const float*)d_in[0];
  const float* mask  = (const float*)d_in[1];
  const float* W_in  = (const float*)d_in[2];
  const float* b_in  = (const float*)d_in[3];
  const float* W_t   = (const float*)d_in[4];
  const float* b_t   = (const float*)d_in[5];
  const float* W_sfc = (const float*)d_in[6];
  const float* b_sfc = (const float*)d_in[7];
  const float* W_f1  = (const float*)d_in[8];
  const float* b_f1  = (const float*)d_in[9];
  const float* W_f2  = (const float*)d_in[10];
  const float* b_f2  = (const float*)d_in[11];
  const float* W_ff  = (const float*)d_in[12];
  const float* b_ff  = (const float*)d_in[13];
  const float* W_q   = (const float*)d_in[14];
  const float* b_q   = (const float*)d_in[15];
  const float* W_k   = (const float*)d_in[16];
  const float* b_k   = (const float*)d_in[17];
  const float* W_v   = (const float*)d_in[18];
  const float* b_v   = (const float*)d_in[19];
  const float* W_o   = (const float*)d_in[20];
  const float* b_o   = (const float*)d_in[21];
  const float* W_co  = (const float*)d_in[22];
  const float* b_co  = (const float*)d_in[23];
  const float* W_ot  = (const float*)d_in[24];
  const float* b_ot  = (const float*)d_in[25];

  // ---- workspace layout ----
  char* ws = (char*)d_ws;
  size_t off = 0;
  auto alloc = [&](size_t bytes) -> char* {
    char* p = ws + off;
    off += (bytes + 255) & ~(size_t)255;
    return p;
  };
  const long long ACT = (long long)MROWS * NFEAT * 4;   // 64MB
  float* F0   = (float*)alloc(ACT);
  float* F1   = (float*)alloc(ACT);
  float* bufA = (float*)alloc(ACT);
  float* bufB = (float*)alloc(ACT);
  float* bufC = (float*)alloc(ACT);
  float* bufD = (float*)alloc(ACT);
  float* FSIN = (float*)alloc((long long)5 * MROWS * 19 * 4);
  float* FT0V = (float*)alloc((long long)MROWS * 19 * 4);
  float* PS   = (float*)alloc(8 * 32 * 256 * 4);
  float* PS2  = (float*)alloc(8 * 32 * 256 * 4);
  float* MU   = (float*)alloc(8 * 256 * 4);
  float* RS   = (float*)alloc(8 * 256 * 4);
  float* KSUM = (float*)alloc(8 * 256 * 4);
  float* Zb   = (float*)alloc((long long)MROWS * 4 * 4);
  float* KVP  = (float*)alloc((long long)32 * 8 * 4096 * 4);
  bf16*  KVT  = (bf16*)alloc((long long)32 * 4096 * 2);
  float* CP   = (float*)alloc((long long)64 * 65536 * 4);
  float* Msm  = (float*)alloc((long long)8 * 65536 * 4);
  bf16*  MTb  = (bf16*)alloc((long long)8 * 65536 * 2);
  bf16*  WTt  = (bf16*)alloc(19 * 256 * 2);
  bf16*  WSFI = (bf16*)alloc(5 * 19 * 256 * 2);
  bf16*  W1S  = (bf16*)alloc((long long)5 * 65536 * 2);
  bf16*  W2S  = (bf16*)alloc((long long)5 * 65536 * 2);
  bf16*  WFF  = (bf16*)alloc((long long)10 * 768 * 256 * 2);
  bf16*  WQb  = (bf16*)alloc((long long)10 * 65536 * 2);
  bf16*  WKb  = (bf16*)alloc((long long)10 * 65536 * 2);
  bf16*  WVb  = (bf16*)alloc((long long)10 * 65536 * 2);
  bf16*  WOb  = (bf16*)alloc((long long)10 * 65536 * 2);
  bf16*  WCOb = (bf16*)alloc((long long)10 * 65536 * 2);
  bf16*  WOUT = (bf16*)alloc(256 * 48 * 2);

  auto prep = [&](const float* src, bf16* dst, int O, int I, int J) {
    int tot = O * I * J;
    k_prep<<<(tot + 255) / 256, 256, 0, stream>>>(src, dst, O, I, J);
  };
  // PLAIN GEMM (kc == K, multiple of 32)
  auto gemmP = [&](const float* A, const bf16* Bw, const float* bias,
                   const float* addp, const float* mp, float* C,
                   int N, int K, int act, int maskMode, long long bStride) {
    dim3 g(MROWS / 128, (N + 63) / 64);
    k_gemm<0><<<g, 256, 0, stream>>>(A, Bw, bias, addp, mp, C, N, K, 0, act,
                                     maskMode, bStride);
  };
  // dilated conv3 GEMM (kc=256, K=768)
  auto gemmC = [&](const float* A, const bf16* Bw, const float* bias, float* C,
                   int dil) {
    dim3 g(MROWS / 128, 4);
    k_gemm<1><<<g, 256, 0, stream>>>(A, Bw, bias, nullptr, nullptr, C, 256, 768,
                                     dil, 1 /*relu*/, 0, 0);
  };
  // K=19 pointwise conv GEMM
  auto gemmS = [&](const float* A, const bf16* Bw, const float* bias, float* C) {
    dim3 g(MROWS / 128, 4);
    k_gemm<2><<<g, 256, 0, stream>>>(A, Bw, bias, nullptr, nullptr, C, 256, 19,
                                     0, 0, 0, 0);
  };

  // ---- weight conversion to bf16 n-major [n][K] ----
  prep(W_t, WTt, 256, 19, 1);
  for (int s = 0; s < 5; s++) {
    prep(W_sfc + (long long)s * 256 * 19, WSFI + (long long)s * 19 * 256, 256, 19, 1);
    prep(W_f1 + (long long)s * 65536, W1S + (long long)s * 65536, 256, 256, 1);
    prep(W_f2 + (long long)s * 65536, W2S + (long long)s * 65536, 256, 256, 1);
  }
  for (int i = 0; i < 10; i++) {
    prep(W_ff + (long long)i * 196608, WFF + (long long)i * 196608, 256, 256, 3);
    prep(W_q + (long long)i * 65536, WQb + (long long)i * 65536, 256, 256, 1);
    prep(W_k + (long long)i * 65536, WKb + (long long)i * 65536, 256, 256, 1);
    prep(W_v + (long long)i * 65536, WVb + (long long)i * 65536, 256, 256, 1);
    prep(W_o + (long long)i * 65536, WOb + (long long)i * 65536, 256, 256, 1);
    prep(W_co + (long long)i * 65536, WCOb + (long long)i * 65536, 256, 256, 1);
  }
  prep(W_ot, WOUT, 48, 256, 1);

  // ---- front end ----
  k_conv_in<<<(MROWS * 19 + 255) / 256, 256, 0, stream>>>(x, W_in, b_in, FSIN, FT0V);
  gemmS(FT0V, WTt, b_t, F0);                                // fst

  int count = 0;
  for (int i = 0; i < 10; i++) {
    if (i & 1) {   // SFI at layers 1,3,5,7,9
      gemmS(FSIN + (long long)count * MROWS * 19, WSFI + (long long)count * 19 * 256,
            b_sfc + count * 256, bufA);
      k_corr_part<<<dim3(16, 8, 8), 256, 0, stream>>>(bufA, F0, CP);
      k_corr_reduce<<<2048, 256, 0, stream>>>(CP, Msm);
      k_softmax<<<2048, 256, 0, stream>>>(Msm);
      k_mt<<<2048, 256, 0, stream>>>(Msm, MTb);
      // cross = ft @ m^T + ft  (per-batch B, n-major m)
      gemmP(F0, MTb, nullptr, F0, nullptr, bufB, 256, 256, 0, 0, 65536);
      gemmP(bufB, W1S + (long long)count * 65536, b_f1 + count * 256,
            nullptr, nullptr, bufC, 256, 256, 3 /*gelu*/, 0, 0);
      gemmP(bufC, W2S + (long long)count * 65536, b_f2 + count * 256,
            F0, mask, F1, 256, 256, 0, 1 /*mask after add*/, 0);
      { float* t = F0; F0 = F1; F1 = t; }
      count++;
    }
    const int dil = 1 << i;
    // out = relu(dilated conv3) as K=768 GEMM
    gemmC(F0, WFF + (long long)i * 196608, b_ff + i * 256, bufA, dil);
    // instance norm over T
    k_stats1<<<256, 256, 0, stream>>>(bufA, PS, PS2);
    k_stats2<<<8, 256, 0, stream>>>(PS, PS2, MU, RS, KSUM, 1);
    k_norm<<<65536, 256, 0, stream>>>(bufA, MU, RS, bufB);
    // projections
    gemmP(bufB, WQb + (long long)i * 65536, b_q + i * 256, nullptr, nullptr, bufC,
          256, 256, 2 /*sigmoid*/, 0, 0);
    gemmP(bufB, WKb + (long long)i * 65536, b_k + i * 256, nullptr, nullptr, bufD,
          256, 256, 2 /*sigmoid*/, 0, 0);
    gemmP(bufB, WVb + (long long)i * 65536, b_v + i * 256, nullptr, nullptr, F1,
          256, 256, 0, 0, 0);              // V lives in F1 (free until final epilogue)
    // k-sum, kv, z, attention
    k_stats1<<<256, 256, 0, stream>>>(bufD, PS, PS2);
    k_stats2<<<8, 256, 0, stream>>>(PS, PS2, MU, RS, KSUM, 0);
    k_kv_part<<<dim3(8, 32), 256, 0, stream>>>(bufD, F1, KVP);
    k_kv_reduce<<<512, 256, 0, stream>>>(KVP, KVT);
    k_z<<<1024, 256, 0, stream>>>(bufC, KSUM, Zb);
    k_qkvz<<<dim3(1024, 4), 256, 0, stream>>>(bufC, KVT, Zb, bufB);
    // a = (att @ Wo + bo)*mask ; AO = a + conv_out
    gemmP(bufB, WOb + (long long)i * 65536, b_o + i * 256, bufA, mask, bufD,
          256, 256, 0, 2 /*mask before add*/, 0);
    // fst = (x + conv1d(AO)) * mask
    gemmP(bufD, WCOb + (long long)i * 65536, b_co + i * 256, F0, mask, F1,
          256, 256, 0, 1 /*mask after add*/, 0);
    { float* t = F0; F0 = F1; F1 = t; }
  }

  // ---- output head: 256 -> 48, then to (B,48,T) with mask ----
  gemmP(F0, WOUT, b_ot, nullptr, nullptr, bufA, 48, 256, 0, 0, 0);
  k_out<<<(MROWS * 48) / 256, 256, 0, stream>>>(bufA, mask, (float*)d_out);
}
